// MVN_DDI_Block_57123065037184
// MI455X (gfx1250) — matrix operand order repaired and checked by reading, not compiled, not run on hardware
//
#include <hip/hip_runtime.h>

#define DD 128          // hidden dim
#define TB 256

typedef __attribute__((ext_vector_type(16))) __bf16 v16bf;
typedef __attribute__((ext_vector_type(8)))  float  v8f;

// ---------------- device helpers ----------------

__device__ __forceinline__ void atomicMaxF(float* addr, float v) {
  // standard IEEE trick: signed-int max for v>=0, unsigned min for v<0
  if (v >= 0.f) atomicMax((int*)addr, __float_as_int(v));
  else          atomicMin((unsigned int*)addr, __float_as_uint(v));
}

__device__ __forceinline__ void atomAddF(float* p, float v) {
  unsafeAtomicAdd(p, v);   // -> global_atomic_add_f32
}

__device__ __forceinline__ float edge_logit(const float* __restrict__ als,
                                            const float* __restrict__ ald,
                                            int s_, int d_, int H, int h) {
  float l = als[(size_t)s_ * H + h] + ald[(size_t)d_ * H + h];
  return l > 0.f ? l : 0.2f * l;   // leaky_relu 0.2
}

// ---------------- WMMA GEMM: C[rows,128] = A[rows,K] @ W[K,128] (+bias) ----------------
// A optionally split at K1 between A and A2 (concat along K). One wave per 16x16 tile,
// 8 waves/block cover the 128 output columns. bf16 inputs, f32 accumulate.
// K templated so the K-loop fully unrolls -> back-to-back v_wmma with pipelined loads.
template <int K, int K1>
__global__ void k_gemm(const float* __restrict__ A, const float* __restrict__ A2,
                       const float* __restrict__ W,      // [K,128] row-major
                       const float* __restrict__ bias,   // [128] or null
                       float* __restrict__ C) {
  const int lane = threadIdx.x & 31;
  const int wv   = threadIdx.x >> 5;        // 0..7 -> column tile
  const int half = lane >> 4;               // 0/1
  const int mn   = lane & 15;
  const int gn   = wv * 16 + mn;            // output column
  const long grow = (long)blockIdx.x * 16 + mn;   // A row for this lane (A fragment)

  const float* __restrict__ Arow  = A + grow * (long)K1;
  const float* __restrict__ A2row = (K > K1) ? (A2 + grow * (long)(K - K1)) : nullptr;

  v8f c = {};
#pragma unroll
  for (int k0 = 0; k0 < K; k0 += 32) {
    const int kA0 = k0 + half * 16;      // A: contiguous, lanes 0-15 K=0..15, lanes 16-31 K=16..31
    const int kA1 = k0 + 8 + half * 16;
    const int kB  = k0 + 16 * half;      // B: lanes 0-15 K=0..15, lanes 16-31 K=16..31

    // 8-element K-groups are 8-aligned and K1 is a multiple of 8, so a group never
    // straddles the A/A2 concat boundary -> two float4 loads per group.
    const float* p0 = (kA0 < K1) ? (Arow + kA0) : (A2row + (kA0 - K1));
    const float* p1 = (kA1 < K1) ? (Arow + kA1) : (A2row + (kA1 - K1));
    float4 a0 = ((const float4*)p0)[0];
    float4 a1 = ((const float4*)p0)[1];
    float4 a2 = ((const float4*)p1)[0];
    float4 a3 = ((const float4*)p1)[1];

    v16bf a, b;
    a[0]  = (__bf16)a0.x; a[1]  = (__bf16)a0.y; a[2]  = (__bf16)a0.z; a[3]  = (__bf16)a0.w;
    a[4]  = (__bf16)a1.x; a[5]  = (__bf16)a1.y; a[6]  = (__bf16)a1.z; a[7]  = (__bf16)a1.w;
    a[8]  = (__bf16)a2.x; a[9]  = (__bf16)a2.y; a[10] = (__bf16)a2.z; a[11] = (__bf16)a2.w;
    a[12] = (__bf16)a3.x; a[13] = (__bf16)a3.y; a[14] = (__bf16)a3.z; a[15] = (__bf16)a3.w;
#pragma unroll
    for (int e = 0; e < 8; ++e) {
      b[e]     = (__bf16)W[(long)(kB + e) * DD + gn];
      b[e + 8] = (__bf16)W[(long)(kB + 8 + e) * DD + gn];
    }
    c = __builtin_amdgcn_wmma_f32_16x16x32_bf16(false, a, false, b, (short)0, c,
                                                false, false);
  }

  const float bv = bias ? bias[gn] : 0.f;   // hoisted: one load, no per-element branch
#pragma unroll
  for (int r = 0; r < 8; ++r) {
    long gm = (long)blockIdx.x * 16 + half * 8 + r;   // C layout: M = r + 8*half
    C[gm * DD + gn] = c[r] + bv;
  }
}

// ---------------- fill / init ----------------

__global__ void k_fill(float* __restrict__ p, float v, long n) {
  long i = (long)blockIdx.x * blockDim.x + threadIdx.x;
  if (i < n) p[i] = v;
}

__global__ void k_init_bias(float* __restrict__ out, const float* __restrict__ bias, long n) {
  long i = (long)blockIdx.x * blockDim.x + threadIdx.x;
  if (i < n) out[i] = bias[i & (DD - 1)];
}

// ---------------- per-node attention logits: al = <x, a> per head ----------------
// wave per node; lane holds 4 contiguous elems -> head = lane / (32/H)
__global__ void k_node_dots(const float* __restrict__ Xs, const float* __restrict__ Xd,
                            const float* __restrict__ avs, const float* __restrict__ avd,
                            int H, float* __restrict__ als, float* __restrict__ ald, int Nn) {
  long t = (long)blockIdx.x * blockDim.x + threadIdx.x;
  int lane = (int)(t & 31);
  long w = t >> 5;
  if (w >= Nn) return;
  float4 xs = ((const float4*)(Xs + w * DD))[lane];
  float4 xd = ((const float4*)(Xd + w * DD))[lane];
  float4 vs = ((const float4*)avs)[lane];
  float4 vd = ((const float4*)avd)[lane];
  float ps = xs.x * vs.x + xs.y * vs.y + xs.z * vs.z + xs.w * vs.w;
  float pd = xd.x * vd.x + xd.y * vd.y + xd.z * vd.z + xd.w * vd.w;
  int width = 32 / H;
  for (int off = width >> 1; off; off >>= 1) {
    ps += __shfl_xor(ps, off, 32);
    pd += __shfl_xor(pd, off, 32);
  }
  if ((lane & (width - 1)) == 0) {
    int h = lane / width;
    als[w * H + h] = ps;
    ald[w * H + h] = pd;
  }
}

// ---------------- edge softmax pipeline ----------------

__global__ void k_edge_max(const int* __restrict__ src, const int* __restrict__ dst,
                           int E_, int Nn, int selfl,
                           const float* __restrict__ als, const float* __restrict__ ald,
                           int H, float* __restrict__ m) {
  long idx = (long)blockIdx.x * blockDim.x + threadIdx.x;
  long tot = ((long)E_ + (selfl ? Nn : 0)) * H;
  if (idx >= tot) return;
  long e = idx / H; int h = (int)(idx - e * H);
  int s_, d_;
  if (e < E_) { s_ = src[e]; d_ = dst[e]; } else { s_ = d_ = (int)(e - E_); }
  atomicMaxF(&m[(size_t)d_ * H + h], edge_logit(als, ald, s_, d_, H, h));
}

__global__ void k_edge_expsum(const int* __restrict__ src, const int* __restrict__ dst,
                              int E_, int Nn, int selfl,
                              const float* __restrict__ als, const float* __restrict__ ald,
                              int H, const float* __restrict__ m,
                              float* __restrict__ ssum, float* __restrict__ wbuf) {
  long idx = (long)blockIdx.x * blockDim.x + threadIdx.x;
  long tot = ((long)E_ + (selfl ? Nn : 0)) * H;
  if (idx >= tot) return;
  long e = idx / H; int h = (int)(idx - e * H);
  int s_, d_;
  if (e < E_) { s_ = src[e]; d_ = dst[e]; } else { s_ = d_ = (int)(e - E_); }
  float l = edge_logit(als, ald, s_, d_, H, h);
  float wexp = __expf(l - m[(size_t)d_ * H + h]);
  wbuf[idx] = wexp;
  atomAddF(&ssum[(size_t)d_ * H + h], wexp);
}

// wave per edge: out[dst] += X[src] * w/(s+eps), per-head coefficient
__global__ void k_edge_scatter(const int* __restrict__ src, const int* __restrict__ dst,
                               int E_, int Nn, int selfl,
                               const float* __restrict__ X, const float* __restrict__ wbuf,
                               const float* __restrict__ ssum, int H,
                               float* __restrict__ out) {
  long t = (long)blockIdx.x * blockDim.x + threadIdx.x;
  int lane = (int)(t & 31);
  long e = t >> 5;
  long tot = (long)E_ + (selfl ? Nn : 0);
  if (e >= tot) return;
  int s_, d_;
  if (e < E_) { s_ = src[e]; d_ = dst[e]; } else { s_ = d_ = (int)(e - E_); }
  int h = (H == 1) ? 0 : (lane >> 4);                 // FH=64 -> head = lane/16
  float coef = wbuf[e * H + h] / (ssum[(size_t)d_ * H + h] + 1e-16f);
  float4 xv = ((const float4*)(X + (size_t)s_ * DD))[lane];
  float* o = out + (size_t)d_ * DD + lane * 4;
  atomAddF(o + 0, xv.x * coef);
  atomAddF(o + 1, xv.y * coef);
  atomAddF(o + 2, xv.z * coef);
  atomAddF(o + 3, xv.w * coef);
}

// ---------------- LayerNorm + ELU (wave per node) ----------------

__global__ void k_ln_elu(const float* __restrict__ X, const float* __restrict__ g,
                         const float* __restrict__ b, float* __restrict__ Y, int Nn) {
  long t = (long)blockIdx.x * blockDim.x + threadIdx.x;
  int lane = (int)(t & 31);
  long w = t >> 5;
  if (w >= Nn) return;
  float4 x = ((const float4*)(X + w * DD))[lane];
  float s = x.x + x.y + x.z + x.w;
  for (int off = 16; off; off >>= 1) s += __shfl_xor(s, off, 32);
  float mean = s * (1.f / DD);
  float d0 = x.x - mean, d1 = x.y - mean, d2 = x.z - mean, d3 = x.w - mean;
  float q = d0 * d0 + d1 * d1 + d2 * d2 + d3 * d3;
  for (int off = 16; off; off >>= 1) q += __shfl_xor(q, off, 32);
  float rs = rsqrtf(q * (1.f / DD) + 1e-5f);
  float4 gv = ((const float4*)g)[lane];
  float4 bv = ((const float4*)b)[lane];
  float y0 = d0 * rs * gv.x + bv.x; y0 = y0 > 0.f ? y0 : expm1f(y0);
  float y1 = d1 * rs * gv.y + bv.y; y1 = y1 > 0.f ? y1 : expm1f(y1);
  float y2 = d2 * rs * gv.z + bv.z; y2 = y2 > 0.f ? y2 : expm1f(y2);
  float y3 = d3 * rs * gv.w + bv.w; y3 = y3 > 0.f ? y3 : expm1f(y3);
  ((float4*)(Y + w * DD))[lane] = make_float4(y0, y1, y2, y3);
}

// ---------------- SAG readout ----------------

// neigh[dst] += X[src], wave per edge
__global__ void k_scatter_plain(const int* __restrict__ src, const int* __restrict__ dst,
                                int E_, const float* __restrict__ X, float* __restrict__ out) {
  long t = (long)blockIdx.x * blockDim.x + threadIdx.x;
  int lane = (int)(t & 31);
  long e = t >> 5;
  if (e >= E_) return;
  int s_ = src[e], d_ = dst[e];
  float4 xv = ((const float4*)(X + (size_t)s_ * DD))[lane];
  float* o = out + (size_t)d_ * DD + lane * 4;
  atomAddF(o + 0, xv.x); atomAddF(o + 1, xv.y);
  atomAddF(o + 2, xv.z); atomAddF(o + 3, xv.w);
}

__global__ void k_attn(const float* __restrict__ Xn, const float* __restrict__ X,
                       const float* __restrict__ wrel, const float* __restrict__ wroot,
                       const float* __restrict__ brel, float* __restrict__ attn, int Nn) {
  long t = (long)blockIdx.x * blockDim.x + threadIdx.x;
  int lane = (int)(t & 31);
  long w = t >> 5;
  if (w >= Nn) return;
  float4 nv = ((const float4*)(Xn + w * DD))[lane];
  float4 xv = ((const float4*)(X + w * DD))[lane];
  float4 rv = ((const float4*)wrel)[lane];
  float4 tv = ((const float4*)wroot)[lane];
  float p = nv.x * rv.x + nv.y * rv.y + nv.z * rv.z + nv.w * rv.w +
            xv.x * tv.x + xv.y * tv.y + xv.z * tv.z + xv.w * tv.w;
  for (int off = 16; off; off >>= 1) p += __shfl_xor(p, off, 32);
  if (lane == 0) attn[w] = p + brel[0];
}

__global__ void k_batch_max(const float* __restrict__ attn, const int* __restrict__ batch,
                            float* __restrict__ mb, int Nn) {
  long i = (long)blockIdx.x * blockDim.x + threadIdx.x;
  if (i < Nn) atomicMaxF(&mb[batch[i]], attn[i]);
}

__global__ void k_batch_expsum(const float* __restrict__ attn, const int* __restrict__ batch,
                               const float* __restrict__ mb, float* __restrict__ sb,
                               float* __restrict__ wn, int Nn) {
  long i = (long)blockIdx.x * blockDim.x + threadIdx.x;
  if (i < Nn) {
    float w = __expf(attn[i] - mb[batch[i]]);
    wn[i] = w;
    atomAddF(&sb[batch[i]], w);
  }
}

__global__ void k_emb_scatter(const float* __restrict__ X, const int* __restrict__ batch,
                              const float* __restrict__ wn, const float* __restrict__ sb,
                              float* __restrict__ emb, int Nn) {
  long t = (long)blockIdx.x * blockDim.x + threadIdx.x;
  int lane = (int)(t & 31);
  long w = t >> 5;
  if (w >= Nn) return;
  int bg = batch[w];
  float coef = wn[w] / (sb[bg] + 1e-16f);
  float4 xv = ((const float4*)(X + w * DD))[lane];
  float* o = emb + (size_t)bg * DD + lane * 4;
  atomAddF(o + 0, xv.x * coef); atomAddF(o + 1, xv.y * coef);
  atomAddF(o + 2, xv.z * coef); atomAddF(o + 3, xv.w * coef);
}

// ---------------- host ----------------

extern "C" void kernel_launch(void* const* d_in, const int* in_sizes, int n_in,
                              void* d_out, int out_size, void* d_ws, size_t ws_size,
                              hipStream_t stream) {
  const float* h_x         = (const float*)d_in[0];
  const float* t_x         = (const float*)d_in[1];
  const float* W_gat       = (const float*)d_in[2];
  const float* a_src_gat   = (const float*)d_in[3];
  const float* a_dst_gat   = (const float*)d_in[4];
  const float* b_gat       = (const float*)d_in[5];
  const float* ln_g        = (const float*)d_in[6];
  const float* ln_b        = (const float*)d_in[7];
  const float* W_intra     = (const float*)d_in[8];
  const float* a_src_intra = (const float*)d_in[9];
  const float* a_dst_intra = (const float*)d_in[10];
  const float* b_intra     = (const float*)d_in[11];
  const float* W_inter_s   = (const float*)d_in[12];
  const float* W_inter_d   = (const float*)d_in[13];
  const float* a_src_inter = (const float*)d_in[14];
  const float* a_dst_inter = (const float*)d_in[15];
  const float* b_inter     = (const float*)d_in[16];
  const float* W_reduce    = (const float*)d_in[17];
  const float* b_reduce    = (const float*)d_in[18];
  const float* w_rel       = (const float*)d_in[19];
  const float* b_rel       = (const float*)d_in[20];
  const float* w_root      = (const float*)d_in[21];
  const int*   h_ei        = (const int*)d_in[22];
  const int*   t_ei        = (const int*)d_in[23];
  const int*   b_ei        = (const int*)d_in[24];
  const int*   h_batch     = (const int*)d_in[25];
  const int*   t_batch     = (const int*)d_in[26];

  const int  N   = in_sizes[25];
  const int  E   = in_sizes[22] / 2;
  const long ND  = (long)N * DD;
  const int  B   = (int)(((long)out_size - 6 * ND) / (2 * DD));
  const long BD  = (long)B * DD;
  const float NEG = -3.0e38f;

  // workspace layout
  float* ws0  = (float*)d_ws;        // scratch projection / neigh
  float* ws1  = ws0 + ND;            // h_e
  float* ws2  = ws1 + ND;            // t_e
  float* ws3  = ws2 + ND;            // second projection (inter dst)
  float* al_s = ws3 + ND;
  float* al_d = al_s + (long)N * 2;
  float* mbuf = al_d + (long)N * 2;
  float* sbuf = mbuf + (long)N * 2;
  float* wbuf = sbuf + (long)N * 2;
  float* attn = wbuf + (long)(E + N) * 2;
  float* wn   = attn + N;
  float* mb   = wn + N;
  float* sb   = mb + B;

  // output regions (tuple order)
  float* o_hf = (float*)d_out;       // h_fused
  float* o_tf = o_hf + ND;           // t_fused
  float* o_he = o_tf + ND;           // h_emb
  float* o_te = o_he + BD;           // t_emb
  float* o_hi = o_te + BD;           // h_intra
  float* o_ti = o_hi + ND;           // t_intra
  float* o_hx = o_ti + ND;           // h_inter
  float* o_tx = o_hx + ND;           // t_inter

  auto nb = [](long n, int bs) { return (unsigned)((n + bs - 1) / bs); };

  auto gat_edges = [&](const int* sA, const int* dA, int Ec, int selfl, int Hh,
                       const float* Xs, float* out) {
    long tot = (long)Ec + (selfl ? N : 0);
    k_fill<<<nb((long)N * Hh, TB), TB, 0, stream>>>(mbuf, NEG, (long)N * Hh);
    k_fill<<<nb((long)N * Hh, TB), TB, 0, stream>>>(sbuf, 0.f, (long)N * Hh);
    k_edge_max<<<nb(tot * Hh, TB), TB, 0, stream>>>(sA, dA, Ec, N, selfl, al_s, al_d, Hh, mbuf);
    k_edge_expsum<<<nb(tot * Hh, TB), TB, 0, stream>>>(sA, dA, Ec, N, selfl, al_s, al_d, Hh,
                                                       mbuf, sbuf, wbuf);
    k_edge_scatter<<<nb(tot * 32, TB), TB, 0, stream>>>(sA, dA, Ec, N, selfl, Xs, wbuf, sbuf,
                                                        Hh, out);
  };

  const unsigned gemm_blocks = (unsigned)(N / 16);
  const unsigned nodew = nb((long)N * 32, TB);

  // ---- feature GAT (h): p = h_x @ W_gat ; softmax over edges+selfloops ; LN+ELU -> h_e
  k_gemm<64, 64><<<gemm_blocks, 256, 0, stream>>>(h_x, nullptr, W_gat, nullptr, ws0);
  k_node_dots<<<nodew, TB, 0, stream>>>(ws0, ws0, a_src_gat, a_dst_gat, 2, al_s, al_d, N);
  k_init_bias<<<nb(ND, TB), TB, 0, stream>>>(ws1, b_gat, ND);
  gat_edges(h_ei, h_ei + E, E, 1, 2, ws0, ws1);
  k_ln_elu<<<nodew, TB, 0, stream>>>(ws1, ln_g, ln_b, ws1, N);

  // ---- feature GAT (t) -> t_e
  k_gemm<64, 64><<<gemm_blocks, 256, 0, stream>>>(t_x, nullptr, W_gat, nullptr, ws0);
  k_node_dots<<<nodew, TB, 0, stream>>>(ws0, ws0, a_src_gat, a_dst_gat, 2, al_s, al_d, N);
  k_init_bias<<<nb(ND, TB), TB, 0, stream>>>(ws2, b_gat, ND);
  gat_edges(t_ei, t_ei + E, E, 1, 2, ws0, ws2);
  k_ln_elu<<<nodew, TB, 0, stream>>>(ws2, ln_g, ln_b, ws2, N);

  // ---- intra GAT (h)
  k_gemm<128, 128><<<gemm_blocks, 256, 0, stream>>>(ws1, nullptr, W_intra, nullptr, ws0);
  k_node_dots<<<nodew, TB, 0, stream>>>(ws0, ws0, a_src_intra, a_dst_intra, 1, al_s, al_d, N);
  k_init_bias<<<nb(ND, TB), TB, 0, stream>>>(o_hi, b_intra, ND);
  gat_edges(h_ei, h_ei + E, E, 1, 1, ws0, o_hi);

  // ---- intra GAT (t)
  k_gemm<128, 128><<<gemm_blocks, 256, 0, stream>>>(ws2, nullptr, W_intra, nullptr, ws0);
  k_node_dots<<<nodew, TB, 0, stream>>>(ws0, ws0, a_src_intra, a_dst_intra, 1, al_s, al_d, N);
  k_init_bias<<<nb(ND, TB), TB, 0, stream>>>(o_ti, b_intra, ND);
  gat_edges(t_ei, t_ei + E, E, 1, 1, ws0, o_ti);

  // ---- inter GAT t_inter: src=h_e via W_src, dst=t_e via W_dst, edges b[0]->b[1]
  k_gemm<128, 128><<<gemm_blocks, 256, 0, stream>>>(ws1, nullptr, W_inter_s, nullptr, ws0);
  k_gemm<128, 128><<<gemm_blocks, 256, 0, stream>>>(ws2, nullptr, W_inter_d, nullptr, ws3);
  k_node_dots<<<nodew, TB, 0, stream>>>(ws0, ws3, a_src_inter, a_dst_inter, 1, al_s, al_d, N);
  k_init_bias<<<nb(ND, TB), TB, 0, stream>>>(o_tx, b_inter, ND);
  gat_edges(b_ei, b_ei + E, E, 0, 1, ws0, o_tx);

  // ---- inter GAT h_inter: src=t_e via W_src, dst=h_e via W_dst, edges b[1]->b[0]
  k_gemm<128, 128><<<gemm_blocks, 256, 0, stream>>>(ws2, nullptr, W_inter_s, nullptr, ws0);
  k_gemm<128, 128><<<gemm_blocks, 256, 0, stream>>>(ws1, nullptr, W_inter_d, nullptr, ws3);
  k_node_dots<<<nodew, TB, 0, stream>>>(ws0, ws3, a_src_inter, a_dst_inter, 1, al_s, al_d, N);
  k_init_bias<<<nb(ND, TB), TB, 0, stream>>>(o_hx, b_inter, ND);
  gat_edges(b_ei + E, b_ei, E, 0, 1, ws0, o_hx);

  // ---- fuse: [intra | inter] @ W_reduce + b_reduce (split-A GEMM, K=256)
  k_gemm<256, 128><<<gemm_blocks, 256, 0, stream>>>(o_hi, o_hx, W_reduce, b_reduce, o_hf);
  k_gemm<256, 128><<<gemm_blocks, 256, 0, stream>>>(o_ti, o_tx, W_reduce, b_reduce, o_tf);

  // ---- SAG readout (h)
  k_fill<<<nb(ND, TB), TB, 0, stream>>>(ws0, 0.f, ND);
  k_scatter_plain<<<nb((long)E * 32, TB), TB, 0, stream>>>(h_ei, h_ei + E, E, o_hf, ws0);
  k_attn<<<nodew, TB, 0, stream>>>(ws0, o_hf, w_rel, w_root, b_rel, attn, N);
  k_fill<<<nb(B, TB), TB, 0, stream>>>(mb, NEG, B);
  k_fill<<<nb(B, TB), TB, 0, stream>>>(sb, 0.f, B);
  k_batch_max<<<nb(N, TB), TB, 0, stream>>>(attn, h_batch, mb, N);
  k_batch_expsum<<<nb(N, TB), TB, 0, stream>>>(attn, h_batch, mb, sb, wn, N);
  k_fill<<<nb(BD, TB), TB, 0, stream>>>(o_he, 0.f, BD);
  k_emb_scatter<<<nodew, TB, 0, stream>>>(o_hf, h_batch, wn, sb, o_he, N);

  // ---- SAG readout (t)
  k_fill<<<nb(ND, TB), TB, 0, stream>>>(ws0, 0.f, ND);
  k_scatter_plain<<<nb((long)E * 32, TB), TB, 0, stream>>>(t_ei, t_ei + E, E, o_tf, ws0);
  k_attn<<<nodew, TB, 0, stream>>>(ws0, o_tf, w_rel, w_root, b_rel, attn, N);
  k_fill<<<nb(B, TB), TB, 0, stream>>>(mb, NEG, B);
  k_fill<<<nb(B, TB), TB, 0, stream>>>(sb, 0.f, B);
  k_batch_max<<<nb(N, TB), TB, 0, stream>>>(attn, t_batch, mb, N);
  k_batch_expsum<<<nb(N, TB), TB, 0, stream>>>(attn, t_batch, mb, sb, wn, N);
  k_fill<<<nb(BD, TB), TB, 0, stream>>>(o_te, 0.f, BD);
  k_emb_scatter<<<nodew, TB, 0, stream>>>(o_tf, t_batch, wn, sb, o_te, N);
}
